// DeepEmbedAttention_60309930770989
// MI455X (gfx1250) — compile-verified, matrix operand order checked by reading, not run on hardware
//
#include <hip/hip_runtime.h>
#include <hip/hip_bf16.h>
#include <stdint.h>

// ---------------------------------------------------------------------------
// Problem constants (match reference)
// ---------------------------------------------------------------------------
#define BB 4
#define TT 2048
#define CC 1024
#define MTOT (BB * TT)          // 8192 rows total
#define EPS 1e-5f
#define LOG2E 1.44269504088896340736f

typedef __attribute__((ext_vector_type(16))) __bf16 v16bf;
typedef __attribute__((ext_vector_type(8)))  float  v8f;
typedef __attribute__((ext_vector_type(4)))  unsigned int v4u;
typedef __attribute__((ext_vector_type(8)))  int v8i;
typedef __attribute__((ext_vector_type(4)))  int v4i;

#define WMMA_BF16(a, b, c) \
  __builtin_amdgcn_wmma_f32_16x16x32_bf16(false, (a), false, (b), (short)0, (c), false, false)

#if defined(__has_builtin)
#if __has_builtin(__builtin_amdgcn_tensor_load_to_lds)
#define HAVE_TDM 1
#endif
#endif
#ifndef HAVE_TDM
#define HAVE_TDM 0
#endif

__device__ __forceinline__ uint16_t f32_to_bf16(float f) {
  uint32_t u = __float_as_uint(f);
  uint32_t r = (u + 0x7FFFu + ((u >> 16) & 1u)) >> 16;   // round-to-nearest-even
  return (uint16_t)r;
}

// Load a 16x32 bf16 fragment (A or B operand) from a row-major tile.
// Per CDNA5 ISA 7.12.2 (16-bit operand layout): lanes 0-15 hold K-chunks
// {0..7,16..23}, lanes 16-31 hold {8..15,24..31}; row index = lane % 16.
// Two 16-byte loads per lane. `ld` in elements; 16B alignment required.
__device__ __forceinline__ v16bf frag_ld(const __bf16* p, int ld) {
  const int lane = threadIdx.x & 31;
  const __bf16* q = p + (size_t)(lane & 15) * ld + ((lane >> 4) << 3);
  union { v16bf v; float4 f[2]; } u;
  u.f[0] = *(const float4*)(q);
  u.f[1] = *(const float4*)(q + 16);
  return u.v;
}

#if HAVE_TDM
// ---------------------------------------------------------------------------
// TDM: DMA a 2D bf16 tile (tile_d1 rows x tile_d0 elems) from global memory
// into LDS at byte offset `lds_off`, inserting a 4-DWORD pad after every
// 16 DWORDs stored (=> 80B LDS row stride for a 32-elem row).
// D# layout per CDNA5 ISA 8.3/8.4. Tracked by TENSORcnt.
// ---------------------------------------------------------------------------
__device__ __forceinline__ void tdm_load_2d(unsigned lds_off, const void* gaddr,
                                            unsigned tile_d0, unsigned tile_d1,
                                            unsigned tensor_d0, unsigned tensor_d1,
                                            unsigned long long d0_stride) {
  unsigned long long ga = (unsigned long long)gaddr;
  v4u g0;
  g0[0] = 1u;                                            // count=1, user mode
  g0[1] = lds_off;                                       // lds_addr (bytes)
  g0[2] = (unsigned)(ga & 0xFFFFFFFFu);                  // global_addr[31:0]
  g0[3] = (unsigned)((ga >> 32) & 0x01FFFFFFu) | (2u << 30);  // addr[56:32]|type=2
  v8i g1;
  // data_size=1(2B) | pad_enable | pad_interval=3(16 DW) | pad_amount=3(4 DW)
  g1[0] = (int)((1u << 16) | (1u << 20) | (3u << 22) | (3u << 25));
  g1[1] = (int)(tensor_d0 << 16);                        // tensor_dim0[15:0] @63:48
  g1[2] = (int)((tensor_d0 >> 16) | (tensor_d1 << 16));  // dim0[31:16] | dim1[15:0]
  g1[3] = (int)((tensor_d1 >> 16) | (tile_d0 << 16));    // dim1[31:16] | tile_dim0
  g1[4] = (int)(tile_d1 & 0xFFFFu);                      // tile_dim1, tile_dim2=0
  g1[5] = (int)(d0_stride & 0xFFFFFFFFull);              // tensor_dim0_stride[31:0]
  g1[6] = (int)((d0_stride >> 32) & 0xFFFFull);          // stride[47:32]
  g1[7] = 0;
  v4i z = {0, 0, 0, 0};
#if __clang_major__ >= 23
  v8i z8 = {0, 0, 0, 0, 0, 0, 0, 0};
  __builtin_amdgcn_tensor_load_to_lds(g0, g1, z, z, z8, 0);
#else
  __builtin_amdgcn_tensor_load_to_lds(g0, g1, z, z, 0);
#endif
}
#endif // HAVE_TDM

// ---------------------------------------------------------------------------
// fp32 -> bf16 conversion (grid-stride)
// ---------------------------------------------------------------------------
__global__ void cvt_f32_bf16(const float* __restrict__ in,
                             uint16_t* __restrict__ out, long n) {
  long i = (long)blockIdx.x * blockDim.x + threadIdx.x;
  long stride = (long)gridDim.x * blockDim.x;
  for (; i * 4 < n; i += stride) {
    float4 v = *(const float4*)(in + i * 4);
    uint16_t* o = out + i * 4;
    o[0] = f32_to_bf16(v.x); o[1] = f32_to_bf16(v.y);
    o[2] = f32_to_bf16(v.z); o[3] = f32_to_bf16(v.w);
  }
}

// ---------------------------------------------------------------------------
// bf16 WMMA GEMM:  C[M,N] = A[M,K] @ W[N,K]^T   (all row-major, fp32 out)
// Block tile 128x128x32, 256 threads = 8 waves (4 M-sub x 2 N-sub),
// each wave computes 32x64 via 2x4 v_wmma_f32_16x16x32_bf16 accumulators.
// Operand tiles staged into LDS by the Tensor Data Mover (double-buffered,
// DMA of buffer p^1 overlaps WMMA on buffer p); falls back to vector loads
// if the TDM builtin is unavailable.
// ---------------------------------------------------------------------------
#define GBM 128
#define GBN 128
#define GBK 32
#define GLD 40                    // padded LDS row stride in bf16 elems (80B)
#define TILE_B (GBM * GLD * 2)    // bytes per staged tile: 10240
#define BUF_B (2 * TILE_B)        // bytes per double-buffer slot: 20480

__global__ __launch_bounds__(256) void gemm_nt_bf16(
    const __bf16* __restrict__ A, const __bf16* __restrict__ W,
    float* __restrict__ Cout, int M, int N, int K) {
  extern __shared__ __bf16 sm[];   // 2 * BUF_B = 40960 bytes dynamic LDS

  const int m0 = blockIdx.y * GBM;
  const int n0 = blockIdx.x * GBN;
  const int tid = threadIdx.x;
  const int lane = tid & 31;
  const int w = tid >> 5;
  const int wm = w & 3;        // 0..3 -> 32-row slab
  const int wn = w >> 2;       // 0..1 -> 64-col slab

  v8f acc[2][4] = {};
  int p = 0;

#if HAVE_TDM
  if (w == 0) {  // wave 0 is the DMA producer (EXEC ignored by tensor ops)
    tdm_load_2d(0u, A + (size_t)m0 * K, GBK, GBM, (unsigned)K, (unsigned)M,
                (unsigned long long)K);
    tdm_load_2d((unsigned)TILE_B, W + (size_t)n0 * K, GBK, GBN, (unsigned)K,
                (unsigned)N, (unsigned long long)K);
  }
#else
  const int lrow = tid >> 1;          // 0..127
  const int lcol = (tid & 1) * 16;    // 0 or 16
#endif

  for (int kk = 0; kk < K; kk += GBK) {
#if HAVE_TDM
    __syncthreads();   // readers of buffer p^1 (prev iter) are done
    if (w == 0) {
      if (kk + GBK < K) {
        const int pn = p ^ 1;
        tdm_load_2d((unsigned)(pn * BUF_B),
                    A + (size_t)m0 * K + (kk + GBK), GBK, GBM,
                    (unsigned)K, (unsigned)M, (unsigned long long)K);
        tdm_load_2d((unsigned)(pn * BUF_B + TILE_B),
                    W + (size_t)n0 * K + (kk + GBK), GBK, GBN,
                    (unsigned)K, (unsigned)N, (unsigned long long)K);
        __builtin_amdgcn_s_wait_tensorcnt((short)2);  // buffer p complete
      } else {
        __builtin_amdgcn_s_wait_tensorcnt((short)0);
      }
    }
    __syncthreads();   // buffer p data visible to all waves
#else
    const __bf16* ag = A + (size_t)(m0 + lrow) * K + kk + lcol;
    const __bf16* wg = W + (size_t)(n0 + lrow) * K + kk + lcol;
    __bf16* as = sm + lrow * GLD;
    __bf16* wsb = sm + (TILE_B / 2) + lrow * GLD;
    *(float4*)(as + lcol)      = *(const float4*)(ag);
    *(float4*)(as + lcol + 8)  = *(const float4*)(ag + 8);
    *(float4*)(wsb + lcol)     = *(const float4*)(wg);
    *(float4*)(wsb + lcol + 8) = *(const float4*)(wg + 8);
    if (kk + GBK < K) {
      __builtin_prefetch(ag + GBK, 0, 1);
      __builtin_prefetch(wg + GBK, 0, 1);
    }
    __syncthreads();
#endif

    const __bf16* As = sm + (size_t)p * (BUF_B / 2);
    const __bf16* Ws = As + (TILE_B / 2);
    v16bf af[2], bf_[4];
#pragma unroll
    for (int mt = 0; mt < 2; ++mt)
      af[mt] = frag_ld(As + (wm * 32 + mt * 16) * GLD, GLD);
#pragma unroll
    for (int nt = 0; nt < 4; ++nt)
      bf_[nt] = frag_ld(Ws + (wn * 64 + nt * 16) * GLD, GLD);
#pragma unroll
    for (int mt = 0; mt < 2; ++mt)
#pragma unroll
      for (int nt = 0; nt < 4; ++nt)
        acc[mt][nt] = WMMA_BF16(af[mt], bf_[nt], acc[mt][nt]);

#if HAVE_TDM
    p ^= 1;
#else
    __syncthreads();
#endif
  }

  // C/D layout: VGPR r -> row (r + 8*(lane/16)), col (lane%16)
  const int lr = lane & 15, hi = lane >> 4;
#pragma unroll
  for (int mt = 0; mt < 2; ++mt) {
#pragma unroll
    for (int nt = 0; nt < 4; ++nt) {
      int rg = m0 + wm * 32 + mt * 16 + hi * 8;
      int cg = n0 + wn * 64 + nt * 16 + lr;
      float* o = Cout + (size_t)rg * N + cg;
#pragma unroll
      for (int r = 0; r < 8; ++r) o[(size_t)r * N] = acc[mt][nt][r];
    }
  }
}

// ---------------------------------------------------------------------------
// Fused token-shift mix + LayerNorm -> bf16.
// MODE 0: q path (no emb).  MODE 1: k path (*emb[idx]).  MODE 2: v path
// (tanh then *emb[idx]).  TRANSPOSED!=0 writes out[b][c][t] (for V).
// One 256-thread block per (b,t) row; 4 channels/thread; LDS tree reduce.
// ---------------------------------------------------------------------------
template <int MODE, int TRANSPOSED>
__global__ __launch_bounds__(256) void mix_ln_kernel(
    const float* __restrict__ raw, const float* __restrict__ emb,
    const int* __restrict__ idx, const float* __restrict__ xmix,
    const float* __restrict__ gamma, const float* __restrict__ beta,
    uint16_t* __restrict__ out) {
  __shared__ float s1[256];
  __shared__ float s2[256];

  const int row = blockIdx.x;          // b*T + t
  const int t = row % TT;
  const int b = row / TT;
  const int tid = threadIdx.x;
  const int c0 = tid * 4;

  float mx[4];
  float lsum = 0.f, lsq = 0.f;
#pragma unroll
  for (int j = 0; j < 4; ++j) {
    const int c = c0 + j;
    float cur = raw[(size_t)row * CC + c];
    float prev = (t > 0) ? raw[(size_t)(row - 1) * CC + c] : 0.f;
    if (MODE == 1) {
      cur *= emb[(size_t)idx[row] * CC + c];
      if (t > 0) prev *= emb[(size_t)idx[row - 1] * CC + c];
    } else if (MODE == 2) {
      cur = tanhf(cur) * emb[(size_t)idx[row] * CC + c];
      if (t > 0) prev = tanhf(prev) * emb[(size_t)idx[row - 1] * CC + c];
    }
    float m = cur + (prev - cur) * xmix[c];
    mx[j] = m;
    lsum += m;
    lsq += m * m;
  }
  s1[tid] = lsum; s2[tid] = lsq;
  __syncthreads();
  for (int off = 128; off > 0; off >>= 1) {
    if (tid < off) { s1[tid] += s1[tid + off]; s2[tid] += s2[tid + off]; }
    __syncthreads();
  }
  const float mean = s1[0] * (1.0f / CC);
  const float var = s2[0] * (1.0f / CC) - mean * mean;
  const float rstd = rsqrtf(var + EPS);
#pragma unroll
  for (int j = 0; j < 4; ++j) {
    const int c = c0 + j;
    float y = (mx[j] - mean) * rstd * gamma[c] + beta[c];
    if (TRANSPOSED)
      out[((size_t)b * CC + c) * TT + t] = f32_to_bf16(y);
    else
      out[(size_t)row * CC + c] = f32_to_bf16(y);
  }
}

// ---------------------------------------------------------------------------
// Attention: per 64-row q-tile. scores = 64*tanh(qk^T/1024) are bounded by
// 64, so softmax uses the constant max 64 => single pass:
//   P = exp2((s-64)*log2e) into LDS (64 x 2056 bf16), row sums l in regs,
//   then O = (P @ V) / l with WMMA (P fragments from LDS, V pre-transposed).
// 128 threads = 4 waves, each wave owns 16 q-rows.
// ---------------------------------------------------------------------------
#define PLD 2056   // padded LDS row stride in bf16 elems (4112B, 16B aligned)

__global__ __launch_bounds__(128) void attn_kernel(
    const __bf16* __restrict__ qln, const __bf16* __restrict__ kln,
    const __bf16* __restrict__ vlnT, float* __restrict__ out) {
  extern __shared__ uint16_t Ps[];                 // 64 * PLD bf16
  float* red = (float*)&Ps[64 * PLD];              // 64 row sums

  const int tiles_per_b = TT / 64;
  const int b = blockIdx.x / tiles_per_b;
  const int i0 = (blockIdx.x % tiles_per_b) * 64;  // q-tile base within batch
  const int tid = threadIdx.x;
  const int lane = tid & 31;
  const int w = tid >> 5;
  const int m0l = w * 16;                          // wave's local row base
  const int lr = lane & 15, hi = lane >> 4;
  const int jmax = i0 + 64;                        // causal key extent

  const __bf16* qb = qln + (size_t)b * TT * CC;
  const __bf16* kb = kln + (size_t)b * TT * CC;
  const __bf16* vb = vlnT + (size_t)b * CC * TT;

  float psum[8];
#pragma unroll
  for (int r = 0; r < 8; ++r) psum[r] = 0.f;

  // ---- Phase 1: scores -> P (LDS) + row sums ----
  for (int jt = 0; jt < jmax; jt += 64) {
    v8f acc[4] = {};
    for (int kk = 0; kk < CC; kk += 32) {
      v16bf a = frag_ld(qb + (size_t)(i0 + m0l) * CC + kk, CC);
#pragma unroll
      for (int nt = 0; nt < 4; ++nt) {
        v16bf bf_ = frag_ld(kb + (size_t)(jt + nt * 16) * CC + kk, CC);
        acc[nt] = WMMA_BF16(a, bf_, acc[nt]);
      }
    }
#pragma unroll
    for (int nt = 0; nt < 4; ++nt) {
      const int j = jt + nt * 16 + lr;
#pragma unroll
      for (int r = 0; r < 8; ++r) {
        const int i = i0 + m0l + r + hi * 8;
        float s = 64.0f * tanhf(acc[nt][r] * (1.0f / 1024.0f));
        float p = (j <= i) ? exp2f((s - 64.0f) * LOG2E) : 0.0f;
        psum[r] += p;
        Ps[(size_t)(m0l + r + hi * 8) * PLD + j] = f32_to_bf16(p);
      }
    }
  }
  // Row-sum reduce across the 16 lanes of each half-wave.
#pragma unroll
  for (int r = 0; r < 8; ++r) {
    float v = psum[r];
    v += __shfl_xor(v, 1, 16);
    v += __shfl_xor(v, 2, 16);
    v += __shfl_xor(v, 4, 16);
    v += __shfl_xor(v, 8, 16);
    if (lr == 0) red[m0l + hi * 8 + r] = v;
  }
  __syncthreads();

  // ---- Phase 2: O = (P @ V) / l ----
  const __bf16* Pb = (const __bf16*)Ps;
  float linv[8];
#pragma unroll
  for (int r = 0; r < 8; ++r) linv[r] = 1.0f / red[m0l + hi * 8 + r];

  for (int nt0 = 0; nt0 < CC; nt0 += 64) {
    v8f acc[4] = {};
    for (int kk = 0; kk < jmax; kk += 32) {
      v16bf a = frag_ld(Pb + (size_t)m0l * PLD + kk, PLD);
#pragma unroll
      for (int nn = 0; nn < 4; ++nn) {
        v16bf bf_ = frag_ld(vb + (size_t)(nt0 + nn * 16) * TT + kk, TT);
        acc[nn] = WMMA_BF16(a, bf_, acc[nn]);
      }
    }
#pragma unroll
    for (int nn = 0; nn < 4; ++nn) {
      const int cg = nt0 + nn * 16 + lr;
#pragma unroll
      for (int r = 0; r < 8; ++r) {
        const int tg = i0 + m0l + r + hi * 8;
        out[((size_t)b * TT + tg) * CC + cg] = acc[nn][r] * linv[r];
      }
    }
  }
}

// ---------------------------------------------------------------------------
// Host-side orchestration
// ---------------------------------------------------------------------------
extern "C" void kernel_launch(void* const* d_in, const int* in_sizes, int n_in,
                              void* d_out, int out_size, void* d_ws, size_t ws_size,
                              hipStream_t stream) {
  (void)in_sizes; (void)n_in; (void)out_size; (void)ws_size;

  const float* x    = (const float*)d_in[0];
  const int*   idx  = (const int*)d_in[1];
  const float* Wq   = (const float*)d_in[2];
  const float* Wk1  = (const float*)d_in[3];
  const float* Wk2  = (const float*)d_in[4];
  const float* Wv1  = (const float*)d_in[5];
  const float* Wv2  = (const float*)d_in[6];
  const float* kemb = (const float*)d_in[7];
  const float* vemb = (const float*)d_in[8];
  const float* x_q  = (const float*)d_in[9];
  const float* x_k  = (const float*)d_in[10];
  const float* x_v  = (const float*)d_in[11];
  const float* g_q  = (const float*)d_in[12];
  const float* b_q  = (const float*)d_in[13];
  const float* g_k  = (const float*)d_in[14];
  const float* b_k  = (const float*)d_in[15];
  const float* g_v  = (const float*)d_in[16];
  const float* b_v  = (const float*)d_in[17];
  float* out = (float*)d_out;

  // Workspace carve-up (bytes)
  char* ws = (char*)d_ws;
  const size_t XB_BYTES  = (size_t)MTOT * CC * 2;      // 16 MB bf16
  const size_t W_BYTES   = (size_t)CC * CC * 2;        // 2 MB bf16 each
  const size_t F32_BYTES = (size_t)MTOT * CC * 4;      // 32 MB fp32

  uint16_t* xb   = (uint16_t*)(ws);
  uint16_t* wqb  = (uint16_t*)(ws + XB_BYTES);
  uint16_t* wk1b = (uint16_t*)(ws + XB_BYTES + 1 * W_BYTES);
  uint16_t* wk2b = (uint16_t*)(ws + XB_BYTES + 2 * W_BYTES);
  uint16_t* wv1b = (uint16_t*)(ws + XB_BYTES + 3 * W_BYTES);
  uint16_t* wv2b = (uint16_t*)(ws + XB_BYTES + 4 * W_BYTES);
  char* p = ws + XB_BYTES + 5 * W_BYTES;
  float*    f32a = (float*)(p);                 p += F32_BYTES;   // GEMM out
  float*    f32b = (float*)(p);                 p += F32_BYTES;   // hidden t
  uint16_t* tb   = (uint16_t*)(p);              p += XB_BYTES;    // bf16(t)
  uint16_t* qln  = (uint16_t*)(p);              p += XB_BYTES;
  uint16_t* kln  = (uint16_t*)(p);              p += XB_BYTES;
  uint16_t* vlnT = (uint16_t*)(p);

  const long NX = (long)MTOT * CC;       // 8.39M
  const long NW = (long)CC * CC;         // 1.05M
  dim3 cvtGrid(2048), cvtBlk(256);
  dim3 gemmGrid(CC / GBN, MTOT / GBM), gemmBlk(256);   // (8, 64)
  dim3 lnGrid(MTOT), lnBlk(256);
  const size_t gemm_lds = 2 * BUF_B;     // 40960 B (double-buffered tiles)

  // Stage bf16 copies of x and weights
  cvt_f32_bf16<<<cvtGrid, cvtBlk, 0, stream>>>(x, xb, NX);
  cvt_f32_bf16<<<cvtGrid, cvtBlk, 0, stream>>>(Wq, wqb, NW);
  cvt_f32_bf16<<<cvtGrid, cvtBlk, 0, stream>>>(Wk1, wk1b, NW);
  cvt_f32_bf16<<<cvtGrid, cvtBlk, 0, stream>>>(Wk2, wk2b, NW);
  cvt_f32_bf16<<<cvtGrid, cvtBlk, 0, stream>>>(Wv1, wv1b, NW);
  cvt_f32_bf16<<<cvtGrid, cvtBlk, 0, stream>>>(Wv2, wv2b, NW);

  // q path
  gemm_nt_bf16<<<gemmGrid, gemmBlk, gemm_lds, stream>>>((const __bf16*)xb, (const __bf16*)wqb, f32a, MTOT, CC, CC);
  mix_ln_kernel<0, 0><<<lnGrid, lnBlk, 0, stream>>>(f32a, nullptr, idx, x_q, g_q, b_q, qln);

  // k path: (x@Wk1^T)@Wk2^T * k_emb[idx]
  gemm_nt_bf16<<<gemmGrid, gemmBlk, gemm_lds, stream>>>((const __bf16*)xb, (const __bf16*)wk1b, f32b, MTOT, CC, CC);
  cvt_f32_bf16<<<cvtGrid, cvtBlk, 0, stream>>>(f32b, tb, NX);
  gemm_nt_bf16<<<gemmGrid, gemmBlk, gemm_lds, stream>>>((const __bf16*)tb, (const __bf16*)wk2b, f32a, MTOT, CC, CC);
  mix_ln_kernel<1, 0><<<lnGrid, lnBlk, 0, stream>>>(f32a, kemb, idx, x_k, g_k, b_k, kln);

  // v path: tanh((x@Wv1^T)@Wv2^T) * v_emb[idx]  -> transposed [b][c][t]
  gemm_nt_bf16<<<gemmGrid, gemmBlk, gemm_lds, stream>>>((const __bf16*)xb, (const __bf16*)wv1b, f32b, MTOT, CC, CC);
  cvt_f32_bf16<<<cvtGrid, cvtBlk, 0, stream>>>(f32b, tb, NX);
  gemm_nt_bf16<<<gemmGrid, gemmBlk, gemm_lds, stream>>>((const __bf16*)tb, (const __bf16*)wv2b, f32a, MTOT, CC, CC);
  mix_ln_kernel<2, 1><<<lnGrid, lnBlk, 0, stream>>>(f32a, vemb, idx, x_v, g_v, b_v, vlnT);

  // Attention: one block per 64-row q-tile, P tile lives in dynamic LDS.
  const size_t attn_lds = (size_t)64 * PLD * 2 + 64 * 4;   // ~257.5 KB < 320 KB
  dim3 attnGrid(BB * (TT / 64)), attnBlk(128);
  attn_kernel<<<attnGrid, attnBlk, attn_lds, stream>>>(
      (const __bf16*)qln, (const __bf16*)kln, (const __bf16*)vlnT, out);
}
